// QuantumConsciousnessNetwork_20306605375891
// MI455X (gfx1250) — matrix operand order repaired and checked by reading, not compiled, not run on hardware
//
#include <hip/hip_runtime.h>
#include <hip/hip_bf16.h>

// ---------------------------------------------------------------------------
// QuantumConsciousnessNetwork forward for MI455X (gfx1250), bf16 WMMA path.
// B=64, L=256, T=500, E=256, H=8, D=32, F=64, C=8, NL=4
// ---------------------------------------------------------------------------

typedef __attribute__((ext_vector_type(16))) __bf16 bf16x16_t;
typedef __attribute__((ext_vector_type(8)))  float  f32x8_t;

union FragAB {
    bf16x16_t v;
    uint4     q[2];
    __device__ FragAB() {}
};

__device__ __forceinline__ f32x8_t fzero8() {
    f32x8_t z;
    for (int i = 0; i < 8; ++i) z[i] = 0.f;
    return z;
}

__device__ __forceinline__ f32x8_t wmma_bf16(const FragAB& a, const FragAB& b, f32x8_t c) {
    return __builtin_amdgcn_wmma_f32_16x16x32_bf16(false, a.v, false, b.v, (short)0, c,
                                                   false, false);
}

__device__ __forceinline__ float sigf(float x) { return 1.f / (1.f + __expf(-x)); }

__device__ __forceinline__ float apply_act(float v, int act) {
    switch (act) {
        case 1: return tanhf(v);
        case 2: return fmaxf(v, 0.f);
        case 3: return sigf(v);
        case 4: return 0.5f * v * (1.f + erff(v * 0.70710678118654752f));  // exact gelu
        default: return v;
    }
}

// ---------------------------------------------------------------------------
// Generic bf16 WMMA GEMM:  C[M,N] = act( A[M,K](bf16) * Bt[N,K](bf16)^T + bias + resid )
// Block = 64 threads (2 waves), tile 64x64; each wave owns a 32x64 sub-tile
// (2 A-fragments share 4 B-fragments -> 8 WMMAs per 32-k step).
// B tile double-buffered in LDS (single barrier per k iteration).
// Requirements: M % 64 == 0, K % 32 == 0.  Bt stored [N][ldb], contiguous in K.
// ---------------------------------------------------------------------------
__global__ __launch_bounds__(64) void gemm_kernel(
    const __bf16* __restrict__ A, int lda,
    const __bf16* __restrict__ Bt, int ldb,
    const float* __restrict__ bias,
    const float* __restrict__ resid, int ldres,
    float* __restrict__ Cf, __bf16* __restrict__ Cb, int ldc,
    int M, int N, int K, int act)
{
    __shared__ __bf16 Bs[2][64][32];

    const int tid  = threadIdx.x;
    const int wave = tid >> 5;      // 0..1
    const int lane = tid & 31;
    const int hl   = lane >> 4;     // lane half (0/1)
    const int ln   = lane & 15;

    const int m0 = blockIdx.y * 64 + wave * 32;   // wave's 32 rows
    const int n0 = blockIdx.x * 64;

    f32x8_t acc[2][4];
    for (int r = 0; r < 2; ++r)
        for (int j = 0; j < 4; ++j) acc[r][j] = fzero8();

    // ---- preload B tile 0 (each thread = one of 64 cols, k 0..7 and 16..23 = 2 uint4) ----
    {
        const int gn = n0 + tid;
        uint4 v0 = make_uint4(0u, 0u, 0u, 0u);
        uint4 v1 = make_uint4(0u, 0u, 0u, 0u);
        if (gn < N) {
            const __bf16* src = Bt + (long)gn * ldb;
            v0 = *(const uint4*)(src);
            v1 = *(const uint4*)(src + 16);
        }
        *(uint4*)(&Bs[0][tid][0])  = v0;
        *(uint4*)(&Bs[0][tid][16]) = v1;
    }
    __syncthreads();

    int parity = 0;
    for (int kk = 0; kk < K; kk += 32) {
        // ---- fetch next B tile into registers (overlaps WMMAs below) ----
        const bool hasNext = (kk + 32 < K);
        uint4 nv0 = make_uint4(0u, 0u, 0u, 0u);
        uint4 nv1 = make_uint4(0u, 0u, 0u, 0u);
        if (hasNext) {
            const int gn = n0 + tid;
            if (gn < N) {
                const __bf16* src = Bt + (long)gn * ldb + kk + 32;
                nv0 = *(const uint4*)(src);
                nv1 = *(const uint4*)(src + 16);
            }
        }
        // ---- 2 A fragments (rows m0..+15, m0+16..+31) ----
        FragAB a0, a1;
        const __bf16* Ar0 = A + (long)(m0 + ln) * lda + kk + hl * 8;
        if (hasNext) __builtin_prefetch(Ar0 + 32, 0, 1);
        a0.q[0] = *(const uint4*)(Ar0);
        a0.q[1] = *(const uint4*)(Ar0 + 16);
        const __bf16* Ar1 = Ar0 + (long)16 * lda;
        a1.q[0] = *(const uint4*)(Ar1);
        a1.q[1] = *(const uint4*)(Ar1 + 16);
        // ---- 4 shared B fragments, 8 WMMAs ----
        for (int j = 0; j < 4; ++j) {
            FragAB b;
            const __bf16* Brow = &Bs[parity][j * 16 + ln][hl * 8];
            b.q[0] = *(const uint4*)(Brow);
            b.q[1] = *(const uint4*)(Brow + 16);
            acc[0][j] = wmma_bf16(a0, b, acc[0][j]);
            acc[1][j] = wmma_bf16(a1, b, acc[1][j]);
        }
        // ---- store next tile into the other buffer, one barrier ----
        if (hasNext) {
            *(uint4*)(&Bs[parity ^ 1][tid][0])  = nv0;
            *(uint4*)(&Bs[parity ^ 1][tid][16]) = nv1;
        }
        __syncthreads();
        parity ^= 1;
    }

    // ---- epilogue: C/D layout row = i + 8*hl, col = ln ----
    for (int r = 0; r < 2; ++r) {
        for (int j = 0; j < 4; ++j) {
            const int col = n0 + j * 16 + ln;
            if (col >= N) continue;
            const float bv = bias ? bias[col] : 0.f;
            for (int i = 0; i < 8; ++i) {
                const int row = m0 + r * 16 + i + hl * 8;
                float v = acc[r][j][i] + bv;
                if (resid) v += resid[(long)row * ldres + col];
                v = apply_act(v, act);
                if (Cf) Cf[(long)row * ldc + col] = v;
                if (Cb) Cb[(long)row * ldc + col] = (__bf16)v;
            }
        }
    }
}

// ---------------------------------------------------------------------------
// Flash attention: one wave per (b, h, 16-query tile).  K-dim D=32.
// Qp/Kb: bf16 [B][L][H*D]; Vt: bf16 [B][H][D][L]; AO out bf16 [B][L][H*D].
// ---------------------------------------------------------------------------
__global__ __launch_bounds__(32) void attn_kernel(
    const __bf16* __restrict__ Qp, const __bf16* __restrict__ Kb,
    const __bf16* __restrict__ Vt, __bf16* __restrict__ AO)
{
    __shared__ __bf16 P[16 * 32];

    const int lane = threadIdx.x;
    const int hl = lane >> 4, ln = lane & 15;
    const int qt = blockIdx.x, h = blockIdx.y, b = blockIdx.z;

    // Q' A-fragment (16 rows x 32 dims)
    FragAB qf;
    const __bf16* qptr = Qp + ((long)(b * 256 + qt * 16 + ln) * 256) + h * 32 + hl * 8;
    qf.q[0] = *(const uint4*)(qptr);
    qf.q[1] = *(const uint4*)(qptr + 16);

    f32x8_t o0 = fzero8(), o1 = fzero8();
    float mrow[8], lrow[8];
    for (int i = 0; i < 8; ++i) { mrow[i] = -1e30f; lrow[i] = 0.f; }

    for (int kt = 0; kt < 256; kt += 32) {
        // K^T B-fragments: col = key, K-dim = head dim (contiguous in Kb)
        FragAB k0f, k1f;
        const __bf16* kp0 = Kb + ((long)(b * 256 + kt + ln) * 256) + h * 32 + hl * 8;
        k0f.q[0] = *(const uint4*)(kp0);
        k0f.q[1] = *(const uint4*)(kp0 + 16);
        const __bf16* kp1 = kp0 + 16 * 256;
        k1f.q[0] = *(const uint4*)(kp1);
        k1f.q[1] = *(const uint4*)(kp1 + 16);

        f32x8_t s0 = wmma_bf16(qf, k0f, fzero8());
        f32x8_t s1 = wmma_bf16(qf, k1f, fzero8());

        // online softmax over this 32-key tile (row reductions across 16 lanes)
        for (int i = 0; i < 8; ++i) {
            float mx = fmaxf(s0[i], s1[i]);
            for (int off = 8; off >= 1; off >>= 1)
                mx = fmaxf(mx, __shfl_xor(mx, off, 16));
            const float mnew  = fmaxf(mrow[i], mx);
            const float alpha = __expf(mrow[i] - mnew);
            const float p0 = __expf(s0[i] - mnew);
            const float p1 = __expf(s1[i] - mnew);
            float ps = p0 + p1;
            for (int off = 8; off >= 1; off >>= 1)
                ps += __shfl_xor(ps, off, 16);
            lrow[i] = lrow[i] * alpha + ps;
            mrow[i] = mnew;
            o0[i] *= alpha;
            o1[i] *= alpha;
            const int prow = i + hl * 8;
            P[prow * 32 + ln]      = (__bf16)p0;
            P[prow * 32 + 16 + ln] = (__bf16)p1;
        }
        __syncthreads();

        // P as A-fragment (16x32) via LDS transpose
        FragAB pf;
        const __bf16* pp = &P[ln * 32 + hl * 8];
        pf.q[0] = *(const uint4*)(pp);
        pf.q[1] = *(const uint4*)(pp + 16);

        // V B-fragments: col = head dim, K-dim = keys (contiguous in Vt)
        FragAB v0f, v1f;
        const __bf16* vp0 = Vt + (((long)(b * 8 + h) * 32 + ln) * 256) + kt + hl * 8;
        v0f.q[0] = *(const uint4*)(vp0);
        v0f.q[1] = *(const uint4*)(vp0 + 16);
        const __bf16* vp1 = vp0 + 16 * 256;
        v1f.q[0] = *(const uint4*)(vp1);
        v1f.q[1] = *(const uint4*)(vp1 + 16);

        o0 = wmma_bf16(pf, v0f, o0);
        o1 = wmma_bf16(pf, v1f, o1);
        __syncthreads();
    }

    for (int i = 0; i < 8; ++i) {
        const int row = qt * 16 + i + hl * 8;
        const float inv = 1.f / lrow[i];
        __bf16* dst = AO + ((long)(b * 256 + row) * 256) + h * 32;
        dst[ln]      = (__bf16)(o0[i] * inv);
        dst[16 + ln] = (__bf16)(o1[i] * inv);
    }
}

// ---------------------------------------------------------------------------
// Forward GRU: 256 sequential steps, one workgroup (32 waves), h in LDS,
// gh = h @ Whh^T via WMMA each step.  gi has bhh_r/bhh_z pre-folded.
// Dynamic LDS: hF 64*256 f32 + hB 64*256 bf16 + gh 64*768 f32 = 288 KB.
// ---------------------------------------------------------------------------
__global__ __launch_bounds__(1024) void gru_fwd_kernel(
    const float* __restrict__ gi,      // [64*256][768], row = b*256 + t
    const __bf16* __restrict__ WhhB,   // [768][256] bf16 (row g, contiguous e)
    const float* __restrict__ bhh_n,   // [256] = bhh[512:768]
    float* __restrict__ hout_f)        // [64][256]
{
    extern __shared__ char smem[];
    float*  hF = (float*)smem;                                  // 64*256
    __bf16* hB = (__bf16*)(smem + 64 * 256 * 4);                // 64*256
    float*  gh = (float*)(smem + 64 * 256 * 4 + 64 * 256 * 2);  // 64*768

    const int tid  = threadIdx.x;
    const int wave = tid >> 5;
    const int lane = tid & 31;
    const int hl = lane >> 4, ln = lane & 15;

    for (int i = tid; i < 64 * 256; i += 1024) { hF[i] = 0.f; hB[i] = (__bf16)0.f; }
    __syncthreads();

    for (int t = 0; t < 256; ++t) {
        // phase 1: gh[64][768] = hB[64][256] @ WhhB^T  (192 tiles over 32 waves)
        for (int j = 0; j < 6; ++j) {
            const int tile = wave * 6 + j;      // 0..191
            const int mt = tile & 3;            // 0..3   (rows)
            const int nt = tile >> 2;           // 0..47  (cols)
            f32x8_t acc = fzero8();
            for (int kk = 0; kk < 256; kk += 32) {
                FragAB a, bfr;
                const __bf16* ap = hB + (mt * 16 + ln) * 256 + kk + hl * 8;
                a.q[0] = *(const uint4*)(ap);
                a.q[1] = *(const uint4*)(ap + 16);
                const __bf16* bp = WhhB + (long)(nt * 16 + ln) * 256 + kk + hl * 8;
                bfr.q[0] = *(const uint4*)(bp);
                bfr.q[1] = *(const uint4*)(bp + 16);
                acc = wmma_bf16(a, bfr, acc);
            }
            for (int i = 0; i < 8; ++i)
                gh[(mt * 16 + i + hl * 8) * 768 + nt * 16 + ln] = acc[i];
        }
        __syncthreads();

        // phase 2: gate math + h update (16384 elements / 1024 threads)
        for (int it = 0; it < 16; ++it) {
            const int idx = tid + it * 1024;
            const int b = idx >> 8, jj = idx & 255;
            const float* grow = gi + ((long)((b << 8) + t)) * 768;
            const float ir  = grow[jj];
            const float iz  = grow[256 + jj];
            const float in_ = grow[512 + jj];
            const float hr = gh[b * 768 + jj];
            const float hz = gh[b * 768 + 256 + jj];
            const float hn = gh[b * 768 + 512 + jj] + bhh_n[jj];
            const float r = sigf(ir + hr);
            const float z = sigf(iz + hz);
            const float n = tanhf(in_ + r * hn);
            const float ho = hF[b * 256 + jj];
            const float hnew = (1.f - z) * n + z * ho;
            hF[b * 256 + jj] = hnew;
            hB[b * 256 + jj] = (__bf16)hnew;
        }
        __syncthreads();
    }
    for (int i = tid; i < 64 * 256; i += 1024) hout_f[i] = hF[i];
}

// ---------------------------------------------------------------------------
// LayerNorm over width 256 (one wave per row, wave32 shfl reductions)
// ---------------------------------------------------------------------------
__global__ __launch_bounds__(256) void ln_kernel(
    const float* __restrict__ X, const float* __restrict__ g, const float* __restrict__ bb,
    const float* __restrict__ resid,
    float* __restrict__ Yf, __bf16* __restrict__ Yb, int M)
{
    const int row  = blockIdx.x * 8 + (threadIdx.x >> 5);
    const int lane = threadIdx.x & 31;
    if (row >= M) return;
    const float* x = X + (long)row * 256;
    float vals[8], s = 0.f, s2 = 0.f;
    for (int i = 0; i < 8; ++i) {
        const float v = x[lane + i * 32];
        vals[i] = v; s += v; s2 += v * v;
    }
    for (int off = 16; off >= 1; off >>= 1) {
        s  += __shfl_xor(s,  off, 32);
        s2 += __shfl_xor(s2, off, 32);
    }
    const float mu = s * (1.f / 256.f);
    const float var = s2 * (1.f / 256.f) - mu * mu;
    const float rs = rsqrtf(var + 1e-5f);
    for (int i = 0; i < 8; ++i) {
        const int c = lane + i * 32;
        float y = (vals[i] - mu) * rs * g[c] + bb[c];
        if (resid) y += resid[(long)row * 256 + c];
        if (Yf) Yf[(long)row * 256 + c] = y;
        if (Yb) Yb[(long)row * 256 + c] = (__bf16)y;
    }
}

// ------------------------- small prep / glue kernels ------------------------

// feats[row][0..499]=returns, [500..999]=volumes, [1000..1023]=0 (K padded)
__global__ void feats_kernel(const float* __restrict__ ret, const float* __restrict__ vol,
                             __bf16* __restrict__ out)
{
    const long idx = (long)blockIdx.x * 256 + threadIdx.x;
    if (idx >= (long)16384 * 1024) return;
    const long row = idx >> 10;
    const int  c   = (int)(idx & 1023);
    float v = 0.f;
    if (c < 500)       v = ret[row * 500 + c];
    else if (c < 1000) v = vol[row * 500 + (c - 500)];
    out[idx] = (__bf16)v;
}

// W [K][N] fp32 -> Bt [N][ldt] bf16 (transpose-convert, zero-pad K..ldt)
__global__ void convT_kernel(const float* __restrict__ W, int K, int N, int ldt,
                             __bf16* __restrict__ out)
{
    const long idx = (long)blockIdx.x * 256 + threadIdx.x;
    if (idx >= (long)N * ldt) return;
    const int n = (int)(idx / ldt);
    const int k = (int)(idx % ldt);
    out[idx] = (k < K) ? (__bf16)W[(long)k * N + n] : (__bf16)0.f;
}

// straight fp32 -> bf16
__global__ void conv_kernel(const float* __restrict__ W, long n, __bf16* __restrict__ out)
{
    const long idx = (long)blockIdx.x * 256 + threadIdx.x;
    if (idx < n) out[idx] = (__bf16)W[idx];
}

// M_h[h][d][e] = ent[h][d][e] + (d==e) * cos^2(phase[h][d]) / sqrt(32)
__global__ void buildM_kernel(const float* __restrict__ phase, const float* __restrict__ ent,
                              float* __restrict__ Mh)
{
    const int idx = blockIdx.x * 256 + threadIdx.x;
    if (idx >= 8 * 32 * 32) return;
    const int e = idx & 31, d = (idx >> 5) & 31, h = idx >> 10;
    float v = ent[idx];
    if (d == e) {
        const float cp = cosf(phase[h * 32 + d]);
        v += cp * cp * 0.17677669529663687f;   // 1/sqrt(32)
    }
    Mh[idx] = v;
}

// WqpT[n=h*32+dout][e] = sum_din Wq[e][h*32+din] * Mh[h][din][dout]  (bf16 out)
__global__ void buildWqT_kernel(const float* __restrict__ Wq, const float* __restrict__ Mh,
                                __bf16* __restrict__ WqpT)
{
    const int idx = blockIdx.x * 256 + threadIdx.x;
    if (idx >= 256 * 256) return;
    const int e = idx & 255, n = idx >> 8;
    const int h = n >> 5, dout = n & 31;
    float s = 0.f;
    for (int din = 0; din < 32; ++din)
        s += Wq[e * 256 + h * 32 + din] * Mh[(h * 32 + din) * 32 + dout];
    WqpT[n * 256 + e] = (__bf16)s;
}

// bqp[n] = sum_din bq[h*32+din] * Mh[h][din][dout]
__global__ void buildBqp_kernel(const float* __restrict__ bq, const float* __restrict__ Mh,
                                float* __restrict__ bqp)
{
    const int n = blockIdx.x * 256 + threadIdx.x;
    if (n >= 256) return;
    const int h = n >> 5, dout = n & 31;
    float s = 0.f;
    for (int din = 0; din < 32; ++din)
        s += bq[h * 32 + din] * Mh[(h * 32 + din) * 32 + dout];
    bqp[n] = s;
}

// V [B][L][H*D] bf16 -> Vt [B][H][D][L] bf16
__global__ void vtrans_kernel(const __bf16* __restrict__ V, __bf16* __restrict__ Vt)
{
    const long idx = (long)blockIdx.x * 256 + threadIdx.x;
    if (idx >= (long)64 * 8 * 32 * 256) return;
    const int l = (int)(idx & 255);
    const int d = (int)((idx >> 8) & 31);
    const int h = (int)((idx >> 13) & 7);
    const int b = (int)(idx >> 16);
    Vt[idx] = V[((long)(b * 256 + l)) * 256 + h * 32 + d];
}

// fold bhh_r / bhh_z into bih (for forward GRU gi GEMM bias)
__global__ void foldbias_kernel(const float* __restrict__ bih, const float* __restrict__ bhh,
                                float* __restrict__ out)
{
    const int j = blockIdx.x * 256 + threadIdx.x;
    if (j >= 768) return;
    out[j] = bih[j] + ((j < 512) ? bhh[j] : 0.f);
}

// one-step backward GRU (h0=0) + build cat64 = [fwd_last | bwd_last] bf16
__global__ void post_gru_kernel(const float* __restrict__ gib, const float* __restrict__ bhhb,
                                const float* __restrict__ fwd_last, __bf16* __restrict__ cat)
{
    const int idx = blockIdx.x * 256 + threadIdx.x;
    if (idx >= 64 * 256) return;
    const int b = idx >> 8, j = idx & 255;
    const float r = sigf(gib[b * 768 + j] + bhhb[j]);
    const float z = sigf(gib[b * 768 + 256 + j] + bhhb[256 + j]);
    const float n = tanhf(gib[b * 768 + 512 + j] + r * bhhb[512 + j]);
    const float h = (1.f - z) * n;
    cat[b * 512 + 256 + j] = (__bf16)h;
    cat[b * 512 + j]       = (__bf16)fwd_last[idx];
}

// final = fwd_last*(1-echo) + mixed*echo
__global__ void final_kernel(const float* __restrict__ fwd, const float* __restrict__ mixed,
                             const float* __restrict__ echo,
                             float* __restrict__ Ff, __bf16* __restrict__ Fb)
{
    const int idx = blockIdx.x * 256 + threadIdx.x;
    if (idx >= 64 * 256) return;
    const float e0 = echo[0];
    const float v = fwd[idx] * (1.f - e0) + mixed[idx] * e0;
    Ff[idx] = v;
    Fb[idx] = (__bf16)v;
}

// catcf = [enhanced(256) | field_state(64)] bf16, ld 320
__global__ void packcf_kernel(const float* __restrict__ enh, const float* __restrict__ fs,
                              __bf16* __restrict__ cat)
{
    const int idx = blockIdx.x * 256 + threadIdx.x;
    if (idx >= 64 * 320) return;
    const int b = idx / 320, c = idx % 320;
    cat[idx] = (c < 256) ? (__bf16)enh[b * 256 + c] : (__bf16)fs[b * 64 + (c - 256)];
}

// ---------------------------------------------------------------------------

extern "C" void kernel_launch(void* const* d_in, const int* in_sizes, int n_in,
                              void* d_out, int out_size, void* d_ws, size_t ws_size,
                              hipStream_t stream)
{
    (void)in_sizes; (void)n_in; (void)out_size; (void)ws_size;
    auto F = [&](int i) { return (const float*)d_in[i]; };
    float* out = (float*)d_out;

    // input indices (setup_inputs insertion order, params flattened in literal order)
    const int IN_RET = 0, IN_VOL = 1, IN_EMB_W = 2, IN_EMB_B = 3, IN_ELN_G = 4, IN_ELN_B = 5;
    const int IN_Q0 = 6;  // per layer: qw,qb,kw,kb,vw,vb,ow,ob,phase,ent
    const int IN_GF_WIH = 46, IN_GF_WHH = 47, IN_GF_BIH = 48, IN_GF_BHH = 49;
    const int IN_GB_WIH = 50, IN_GB_BIH = 52, IN_GB_BHH = 53;
    const int IN_MIX_W = 54, IN_MIX_B = 55, IN_MLN_G = 56, IN_MLN_B = 57, IN_ECHO = 58;
    const int IN_FP1_W = 59, IN_FP1_B = 60, IN_FP2_W = 61, IN_FP2_B = 62;
    const int IN_CO1_W = 63, IN_CO1_B = 64, IN_CO2_W = 65, IN_CO2_B = 66;
    const int IN_PR1_W = 67, IN_PR1_B = 68, IN_PR2_W = 69, IN_PR2_B = 70;
    const int IN_RV_W = 71, IN_RV_B = 72, IN_RVS_W = 73, IN_RVS_B = 74;
    const int IN_FU_W = 75, IN_FU_B = 76, IN_FLN_G = 77, IN_FLN_B = 78;
    const int IN_R1_W = 79, IN_R1_B = 80, IN_R2_W = 81, IN_R2_B = 82;
    const int IN_E1_W = 83, IN_E1_B = 84, IN_E2_W = 85, IN_E2_B = 86;
    const int IN_C1_W = 87, IN_C1_B = 88, IN_C2_W = 89, IN_C2_B = 90;

    // ---- workspace bump allocator ----
    size_t off = 0;
    auto alloc = [&](size_t bytes) -> char* {
        char* p = (char*)d_ws + off;
        off += (bytes + 255) & ~(size_t)255;
        return p;
    };
    const long MBL = 16384;  // B*L

    char* region0 = alloc((size_t)MBL * 768 * 4);           // feats(bf16, 32MB) / gi(f32, 48MB) union
    __bf16* featsBF = (__bf16*)region0;
    float*  giF     = (float*)region0;
    float*  xraw  = (float*)alloc((size_t)MBL * 256 * 4);
    float*  xF0   = (float*)alloc((size_t)MBL * 256 * 4);
    float*  xF1   = (float*)alloc((size_t)MBL * 256 * 4);
    __bf16* xB0   = (__bf16*)alloc((size_t)MBL * 256 * 2);
    __bf16* xB1   = (__bf16*)alloc((size_t)MBL * 256 * 2);
    __bf16* QpBF  = (__bf16*)alloc((size_t)MBL * 256 * 2);
    __bf16* KBF   = (__bf16*)alloc((size_t)MBL * 256 * 2);
    __bf16* VBF   = (__bf16*)alloc((size_t)MBL * 256 * 2);
    __bf16* VtBF  = (__bf16*)alloc((size_t)MBL * 256 * 2);
    __bf16* AOBF  = (__bf16*)alloc((size_t)MBL * 256 * 2);

    __bf16* embT  = (__bf16*)alloc(256 * 1024 * 2);
    float*  Mh    = (float*)alloc(8 * 32 * 32 * 4);
    __bf16* WqpT[4]; float* bqp[4]; __bf16* WkT[4]; __bf16* WvT[4]; __bf16* WoT[4];
    for (int li = 0; li < 4; ++li) {
        WqpT[li] = (__bf16*)alloc(256 * 256 * 2);
        bqp[li]  = (float*)alloc(256 * 4);
        WkT[li]  = (__bf16*)alloc(256 * 256 * 2);
        WvT[li]  = (__bf16*)alloc(256 * 256 * 2);
        WoT[li]  = (__bf16*)alloc(256 * 256 * 2);
    }
    __bf16* wihF = (__bf16*)alloc(768 * 256 * 2);
    __bf16* whhF = (__bf16*)alloc(768 * 256 * 2);
    __bf16* wihB = (__bf16*)alloc(768 * 256 * 2);
    float*  biasF = (float*)alloc(768 * 4);
    __bf16* mixT  = (__bf16*)alloc(256 * 512 * 2);
    __bf16* rvT   = (__bf16*)alloc((size_t)8 * 256 * 256 * 2);
    __bf16* rvsT  = (__bf16*)alloc(8 * 256 * 2);
    __bf16* fuseT = (__bf16*)alloc((size_t)256 * 2048 * 2);
    __bf16* ret1T = (__bf16*)alloc(256 * 256 * 2);
    __bf16* ret2T = (__bf16*)alloc(500 * 256 * 2);
    __bf16* ev1T  = (__bf16*)alloc(128 * 256 * 2);
    __bf16* ev2T  = (__bf16*)alloc(3 * 128 * 2);
    __bf16* fp1T  = (__bf16*)alloc(128 * 256 * 2);
    __bf16* fp2T  = (__bf16*)alloc(64 * 128 * 2);
    __bf16* co1T  = (__bf16*)alloc(32 * 64 * 2);
    __bf16* co2T  = (__bf16*)alloc(1 * 32 * 2);
    __bf16* pr1T  = (__bf16*)alloc(32 * 64 * 2);
    __bf16* pr2T  = (__bf16*)alloc(1 * 32 * 2);
    __bf16* cf1T  = (__bf16*)alloc(64 * 320 * 2);
    __bf16* cf2T  = (__bf16*)alloc(1 * 64 * 2);

    float*  gibF   = (float*)alloc(64 * 768 * 4);
    float*  fwdF   = (float*)alloc(64 * 256 * 4);
    __bf16* cat64  = (__bf16*)alloc(64 * 512 * 2);
    float*  mixraw = (float*)alloc(64 * 256 * 4);
    float*  mixedF = (float*)alloc(64 * 256 * 4);
    float*  finalF = (float*)alloc(64 * 256 * 4);
    __bf16* finalB = (__bf16*)alloc(64 * 256 * 2);
    __bf16* t128B  = (__bf16*)alloc(64 * 128 * 2);
    float*  fsF    = (float*)alloc(64 * 64 * 4);
    __bf16* fsB    = (__bf16*)alloc(64 * 64 * 2);
    __bf16* t32aB  = (__bf16*)alloc(64 * 32 * 2);
    __bf16* t32bB  = (__bf16*)alloc(64 * 32 * 2);
    __bf16* viewB  = (__bf16*)alloc((size_t)64 * 2048 * 2);
    float*  fuseR  = (float*)alloc(64 * 256 * 4);
    float*  enhF   = (float*)alloc(64 * 256 * 4);
    __bf16* enhB   = (__bf16*)alloc(64 * 256 * 2);
    __bf16* r1B    = (__bf16*)alloc(64 * 256 * 2);
    __bf16* e1B    = (__bf16*)alloc(64 * 128 * 2);
    __bf16* catCF  = (__bf16*)alloc(64 * 320 * 2);
    __bf16* c64B   = (__bf16*)alloc(64 * 64 * 2);

    auto gblk = [](long n) { return (unsigned)((n + 255) / 256); };
    auto gemm = [&](const __bf16* A, int lda, const __bf16* Bt, int ldb,
                    const float* bias, const float* resid, int ldres,
                    float* Cf, __bf16* Cb, int ldc, int M, int N, int K, int act) {
        dim3 g((N + 63) / 64, M / 64);
        gemm_kernel<<<g, 64, 0, stream>>>(A, lda, Bt, ldb, bias, resid, ldres,
                                          Cf, Cb, ldc, M, N, K, act);
    };

    // ================== weight conversions ==================
    feats_kernel<<<gblk(MBL * 1024), 256, 0, stream>>>(F(IN_RET), F(IN_VOL), featsBF);
    convT_kernel<<<gblk(256 * 1024), 256, 0, stream>>>(F(IN_EMB_W), 1000, 256, 1024, embT);
    for (int li = 0; li < 4; ++li) {
        const int base = IN_Q0 + li * 10;
        buildM_kernel<<<gblk(8192), 256, 0, stream>>>(F(base + 8), F(base + 9), Mh);
        buildWqT_kernel<<<gblk(65536), 256, 0, stream>>>(F(base + 0), Mh, WqpT[li]);
        buildBqp_kernel<<<1, 256, 0, stream>>>(F(base + 1), Mh, bqp[li]);
        convT_kernel<<<gblk(65536), 256, 0, stream>>>(F(base + 2), 256, 256, 256, WkT[li]);
        convT_kernel<<<gblk(65536), 256, 0, stream>>>(F(base + 4), 256, 256, 256, WvT[li]);
        convT_kernel<<<gblk(65536), 256, 0, stream>>>(F(base + 6), 256, 256, 256, WoT[li]);
    }
    conv_kernel<<<gblk(196608), 256, 0, stream>>>(F(IN_GF_WIH), 196608, wihF);
    conv_kernel<<<gblk(196608), 256, 0, stream>>>(F(IN_GF_WHH), 196608, whhF);
    conv_kernel<<<gblk(196608), 256, 0, stream>>>(F(IN_GB_WIH), 196608, wihB);
    foldbias_kernel<<<3, 256, 0, stream>>>(F(IN_GF_BIH), F(IN_GF_BHH), biasF);
    convT_kernel<<<gblk(256 * 512), 256, 0, stream>>>(F(IN_MIX_W), 512, 256, 512, mixT);
    for (int c = 0; c < 8; ++c)
        convT_kernel<<<gblk(65536), 256, 0, stream>>>(F(IN_RV_W) + c * 65536, 256, 256, 256,
                                                      rvT + (size_t)c * 65536);
    conv_kernel<<<gblk(2048), 256, 0, stream>>>(F(IN_RVS_W), 2048, rvsT);
    convT_kernel<<<gblk(256 * 2048), 256, 0, stream>>>(F(IN_FU_W), 2048, 256, 2048, fuseT);
    convT_kernel<<<gblk(65536), 256, 0, stream>>>(F(IN_R1_W), 256, 256, 256, ret1T);
    convT_kernel<<<gblk(500 * 256), 256, 0, stream>>>(F(IN_R2_W), 256, 500, 256, ret2T);
    convT_kernel<<<gblk(128 * 256), 256, 0, stream>>>(F(IN_E1_W), 256, 128, 256, ev1T);
    convT_kernel<<<gblk(3 * 128), 256, 0, stream>>>(F(IN_E2_W), 128, 3, 128, ev2T);
    convT_kernel<<<gblk(128 * 256), 256, 0, stream>>>(F(IN_FP1_W), 256, 128, 256, fp1T);
    convT_kernel<<<gblk(64 * 128), 256, 0, stream>>>(F(IN_FP2_W), 128, 64, 128, fp2T);
    convT_kernel<<<gblk(32 * 64), 256, 0, stream>>>(F(IN_CO1_W), 64, 32, 64, co1T);
    convT_kernel<<<gblk(32), 256, 0, stream>>>(F(IN_CO2_W), 32, 1, 32, co2T);
    convT_kernel<<<gblk(32 * 64), 256, 0, stream>>>(F(IN_PR1_W), 64, 32, 64, pr1T);
    convT_kernel<<<gblk(32), 256, 0, stream>>>(F(IN_PR2_W), 32, 1, 32, pr2T);
    convT_kernel<<<gblk(64 * 320), 256, 0, stream>>>(F(IN_C1_W), 320, 64, 320, cf1T);
    convT_kernel<<<gblk(64), 256, 0, stream>>>(F(IN_C2_W), 64, 1, 64, cf2T);

    // ================== embedding + LN ==================
    gemm(featsBF, 1024, embT, 1024, F(IN_EMB_B), nullptr, 0,
         xraw, nullptr, 256, (int)MBL, 256, 1024, 0);
    ln_kernel<<<MBL / 8, 256, 0, stream>>>(xraw, F(IN_ELN_G), F(IN_ELN_B), nullptr,
                                           xF0, xB0, (int)MBL);

    // ================== 4 quantum-attention layers ==================
    float*  xFc[2] = {xF0, xF1};
    __bf16* xBc[2] = {xB0, xB1};
    int p = 0;
    for (int li = 0; li < 4; ++li) {
        const int base = IN_Q0 + li * 10;
        const int np = 1 - p;
        gemm(xBc[p], 256, WqpT[li], 256, bqp[li], nullptr, 0,
             nullptr, QpBF, 256, (int)MBL, 256, 256, 0);
        gemm(xBc[p], 256, WkT[li], 256, F(base + 3), nullptr, 0,
             nullptr, KBF, 256, (int)MBL, 256, 256, 0);
        gemm(xBc[p], 256, WvT[li], 256, F(base + 5), nullptr, 0,
             nullptr, VBF, 256, (int)MBL, 256, 256, 0);
        vtrans_kernel<<<gblk(MBL * 256), 256, 0, stream>>>(VBF, VtBF);
        attn_kernel<<<dim3(16, 8, 64), 32, 0, stream>>>(QpBF, KBF, VtBF, AOBF);
        gemm(AOBF, 256, WoT[li], 256, F(base + 7), xFc[p], 256,
             xFc[np], xBc[np], 256, (int)MBL, 256, 256, 0);
        p = np;
    }

    // ================== bidirectional GRU (fwd full, bwd 1 step) ==================
    gemm(xBc[p], 256, wihF, 256, biasF, nullptr, 0,
         giF, nullptr, 768, (int)MBL, 768, 256, 0);
    gru_fwd_kernel<<<1, 1024, 294912, stream>>>(giF, whhF, F(IN_GF_BHH) + 512, fwdF);
    // backward: only x[:, L-1] matters -> one GEMM (strided A rows) + one-step combine
    gemm(xBc[p] + 255 * 256, 256 * 256, wihB, 256, F(IN_GB_BIH), nullptr, 0,
         gibF, nullptr, 768, 64, 768, 256, 0);
    post_gru_kernel<<<64, 256, 0, stream>>>(gibF, F(IN_GB_BHH), fwdF, cat64);

    // mixer (last position only) + echo blend
    gemm(cat64, 512, mixT, 512, F(IN_MIX_B), nullptr, 0,
         mixraw, nullptr, 256, 64, 256, 512, 0);
    ln_kernel<<<8, 256, 0, stream>>>(mixraw, F(IN_MLN_G), F(IN_MLN_B), nullptr,
                                     mixedF, nullptr, 64);
    final_kernel<<<64, 256, 0, stream>>>(fwdF, mixedF, F(IN_ECHO), finalF, finalB);

    // ================== heads ==================
    // field_state
    gemm(finalB, 256, fp1T, 256, F(IN_FP1_B), nullptr, 0,
         nullptr, t128B, 128, 64, 128, 256, 1);
    gemm(t128B, 128, fp2T, 128, F(IN_FP2_B), nullptr, 0,
         fsF, fsB, 64, 64, 64, 128, 0);
    // coherence -> out[32192:32256]
    gemm(fsB, 64, co1T, 64, F(IN_CO1_B), nullptr, 0,
         nullptr, t32aB, 32, 64, 32, 64, 2);
    gemm(t32aB, 32, co2T, 32, F(IN_CO2_B), nullptr, 0,
         out + 32192, nullptr, 1, 64, 1, 32, 3);
    // presentiment -> out[32256:32320]
    gemm(fsB, 64, pr1T, 64, F(IN_PR1_B), nullptr, 0,
         nullptr, t32bB, 32, 64, 32, 64, 1);
    gemm(t32bB, 32, pr2T, 32, F(IN_PR2_B), nullptr, 0,
         out + 32256, nullptr, 1, 64, 1, 32, 1);
    // remote viewing (8 heads, tanh) packed into [64][2048]
    for (int c = 0; c < 8; ++c)
        gemm(finalB, 256, rvT + (size_t)c * 65536, 256, F(IN_RV_B) + c * 256, nullptr, 0,
             nullptr, viewB + c * 256, 2048, 64, 256, 256, 1);
    // strengths -> out[32320:32832]
    gemm(finalB, 256, rvsT, 256, F(IN_RVS_B), nullptr, 0,
         out + 32320, nullptr, 8, 64, 8, 256, 3);
    // fuse + LN + residual(final)
    gemm(viewB, 2048, fuseT, 2048, F(IN_FU_B), nullptr, 0,
         fuseR, nullptr, 256, 64, 256, 2048, 0);
    ln_kernel<<<8, 256, 0, stream>>>(fuseR, F(IN_FLN_G), F(IN_FLN_B), finalF,
                                     enhF, enhB, 64);
    // return_pred -> out[0:32000]
    gemm(enhB, 256, ret1T, 256, F(IN_R1_B), nullptr, 0,
         nullptr, r1B, 256, 64, 256, 256, 4);
    gemm(r1B, 256, ret2T, 256, F(IN_R2_B), nullptr, 0,
         out, nullptr, 500, 64, 500, 256, 0);
    // event_pred -> out[32000:32192]
    gemm(enhB, 256, ev1T, 256, F(IN_E1_B), nullptr, 0,
         nullptr, e1B, 128, 64, 128, 256, 4);
    gemm(e1B, 128, ev2T, 128, F(IN_E2_B), nullptr, 0,
         out + 32000, nullptr, 3, 64, 3, 128, 0);
    // confidence -> out[32832:32896]
    packcf_kernel<<<gblk(64 * 320), 256, 0, stream>>>(enhF, fsF, catCF);
    gemm(catCF, 320, cf1T, 320, F(IN_C1_B), nullptr, 0,
         nullptr, c64B, 64, 64, 64, 320, 2);
    gemm(c64B, 64, cf2T, 64, F(IN_C2_B), nullptr, 0,
         out + 32832, nullptr, 1, 64, 1, 64, 3);
}
